// TSHMEncoder_64106681860637
// MI455X (gfx1250) — compile-verified, matrix operand order checked by reading, not run on hardware
//
#include <hip/hip_runtime.h>
#include <hip/hip_bf16.h>
#include <math.h>

// Problem constants (fixed by reference setup_inputs)
#define BSZ   8
#define LSEQ  4096
#define BD    512     // d
#define RR    32      // r
#define KK    8       // K
#define FF    512     // ff
#define BL    (BSZ*LSEQ)

typedef __attribute__((ext_vector_type(2)))  float  v2f;
typedef __attribute__((ext_vector_type(8)))  float  v8f;
typedef __attribute__((ext_vector_type(16))) __bf16 v16bf;
typedef unsigned int uint32x4 __attribute__((ext_vector_type(4)));
typedef int          int32x4  __attribute__((ext_vector_type(4)));
typedef int          int32x8  __attribute__((ext_vector_type(8)));

union V16U8 { v16bf v; unsigned int u[8]; };
union V8FU  { v8f v; float f[8]; };

__device__ __forceinline__ unsigned short f2bf(float f) {
    unsigned int u = __float_as_uint(f);
    unsigned int r = u + 0x7FFFu + ((u >> 16) & 1u);   // round-to-nearest-even
    return (unsigned short)(r >> 16);
}

// ---------------------------------------------------------------- K0: W1/W2 -> bf16
__global__ __launch_bounds__(256) void k0_cvt(const float* __restrict__ W1,
                                              const float* __restrict__ W2,
                                              unsigned short* __restrict__ W1b,
                                              unsigned short* __restrict__ W2b) {
    int t = blockIdx.x * 256 + threadIdx.x;     // 512*512 elements
    W1b[t] = f2bf(W1[t]);
    W2b[t] = f2bf(W2[t]);
}

// ---------------------------------------------------------------- K1: LN, phi, P/Q/gate-dots
// one wave (32 lanes) per (b,l) row; 8 rows per block
__global__ __launch_bounds__(256) void k1_ln_proj(const float* __restrict__ X,
                                                  const float* __restrict__ Uw,
                                                  const float* __restrict__ Vw,
                                                  const float* __restrict__ gw,
                                                  const float* __restrict__ lng,
                                                  const float* __restrict__ lnb,
                                                  float* __restrict__ P,
                                                  float* __restrict__ Q,
                                                  float* __restrict__ cj) {
    __shared__ float sphi[8][BD];
    __shared__ float sxn[8][BD];
    int wave = threadIdx.x >> 5, lane = threadIdx.x & 31;
    long row = (long)blockIdx.x * 8 + wave;     // row in [0, B*L)
    const float* xr = X + row * BD;

    float x[16]; float s = 0.f;
#pragma unroll
    for (int j = 0; j < 16; ++j) { x[j] = xr[lane + 32 * j]; s += x[j]; }
#pragma unroll
    for (int m = 16; m; m >>= 1) s += __shfl_xor(s, m, 32);
    float mean = s * (1.0f / BD);
    float v = 0.f;
#pragma unroll
    for (int j = 0; j < 16; ++j) { float t = x[j] - mean; v += t * t; }
#pragma unroll
    for (int m = 16; m; m >>= 1) v += __shfl_xor(v, m, 32);
    float rstd = rsqrtf(v * (1.0f / BD) + 1e-5f);
#pragma unroll
    for (int j = 0; j < 16; ++j) {
        int dd = lane + 32 * j;
        float xn = (x[j] - mean) * rstd * lng[dd] + lnb[dd];
        sxn[wave][dd] = xn;
        sphi[wave][dd] = tanhf(xn);
    }
    __syncthreads();

    // lane r: P[r] = phi . Uw[r,:], Q[r] = phi . Vw[r,:]
    float p = 0.f, q = 0.f;
    const float* ur = Uw + lane * BD;
    const float* vr = Vw + lane * BD;
    for (int dd = 0; dd < BD; ++dd) {
        float ph = sphi[wave][dd];
        p = fmaf(ph, ur[dd], p);
        q = fmaf(ph, vr[dd], q);
    }
    P[row * RR + lane] = p;
    Q[row * RR + lane] = q;
    if (lane < 3) {   // gate conv taps: cj[j] = Xn . gate_w[:, j]
        float cc = 0.f;
        for (int dd = 0; dd < BD; ++dd) cc = fmaf(sxn[wave][dd], gw[dd * 3 + lane], cc);
        cj[row * 3 + lane] = cc;
    }
}

// ---------------------------------------------------------------- K2a: prefix cumsum over L
__global__ void k2a_cumsum(const float* __restrict__ P, float* __restrict__ S) {
    int b = blockIdx.x, r = threadIdx.x;
    long base = (long)b * LSEQ * RR + r;
    float s = 0.f;
    for (int l = 0; l < LSEQ; ++l) { s += P[base + (long)l * RR]; S[base + (long)l * RR] = s; }
}

// ---------------------------------------------------------------- K2b: e/G -> M, N  (per row)
__global__ __launch_bounds__(256) void k2b_gmn(const float* __restrict__ S,
                                               const float* __restrict__ Q,
                                               const float* __restrict__ Aw,
                                               const float* __restrict__ Ab,
                                               const float* __restrict__ Bw,
                                               const float* __restrict__ Bb,
                                               const float* __restrict__ cv,
                                               float* __restrict__ M,
                                               float* __restrict__ N) {
    long row = (long)blockIdx.x * 256 + threadIdx.x;
    const float* sr = S + row * RR;
    const float* qr = Q + row * RR;
    float sv[RR], qv[RR];
#pragma unroll
    for (int r = 0; r < RR; ++r) { sv[r] = sr[r]; qv[r] = qr[r]; }
    float G[KK];
#pragma unroll
    for (int k = 0; k < KK; ++k) {
        float a = Ab[k] + Bb[k] + cv[k];
        for (int r = 0; r < RR; ++r)
            a = fmaf(Aw[k * RR + r], sv[r], fmaf(Bw[k * RR + r], qv[r], a));
        G[k] = 0.001f * tanhf(a);     // const = 0.1 * 0.1^2
    }
    for (int r = 0; r < RR; ++r) {
        float m = 0.f, n = 0.f;
#pragma unroll
        for (int k = 0; k < KK; ++k) { m = fmaf(G[k], Aw[k * RR + r], m); n = fmaf(G[k], Bw[k * RR + r], n); }
        M[row * RR + r] = m;
        N[row * RR + r] = n;
    }
}

// ---------------------------------------------------------------- K2c: suffix cumsum over L
__global__ void k2c_sufsum(const float* __restrict__ M, float* __restrict__ Ms) {
    int b = blockIdx.x, r = threadIdx.x;
    long base = (long)b * LSEQ * RR + r;
    float s = 0.f;
    for (int l = LSEQ - 1; l >= 0; --l) { s += M[base + (long)l * RR]; Ms[base + (long)l * RR] = s; }
}

// ---------------------------------------------------------------- K2g: gate
__global__ __launch_bounds__(256) void k2g_gate(const float* __restrict__ cj,
                                                const float* __restrict__ gate_b,
                                                float* __restrict__ g) {
    long t = (long)blockIdx.x * 256 + threadIdx.x;   // t over B*L
    int l = (int)(t & (LSEQ - 1));
    float acc = gate_b[0] + cj[t * 3 + 1];
    if (l > 0)        acc += cj[(t - 1) * 3 + 0];
    if (l < LSEQ - 1) acc += cj[(t + 1) * 3 + 2];
    float sg = 1.f / (1.f + expf(-acc));
    g[t] = 1e-6f + (1.f - 1e-6f) * sg;
}

// ---------------------------------------------------------------- K3: grad = Msuf@U + N@V ; X1
// f32 WMMA 16x16x4: 16-row tile per block, 8 waves sweep 32 column tiles
__global__ __launch_bounds__(256) void k3_grad_x1(const float* __restrict__ Ms,
                                                  const float* __restrict__ Nm,
                                                  const float* __restrict__ Uw,
                                                  const float* __restrict__ Vw,
                                                  const float* __restrict__ X,
                                                  const float* __restrict__ g,
                                                  const float* __restrict__ res_scale,
                                                  const float* __restrict__ lscale,
                                                  float* __restrict__ X1) {
    int wave = threadIdx.x >> 5, lane = threadIdx.x & 31;
    int hi = lane >> 4, lanelo = lane & 15;
    long row0 = (long)blockIdx.x * 16;
    int koff = hi * 2;

    // A fragments (ISA: 32-bit A 16x4 — lanes0-15 K0/K1, lanes16-31 K2/K3)
    v2f aM[8], aN[8];
    long arow = row0 + lanelo;
#pragma unroll
    for (int kk = 0; kk < 8; ++kk) {
        aM[kk].x = Ms[arow * RR + kk * 4 + koff];
        aM[kk].y = Ms[arow * RR + kk * 4 + koff + 1];
        aN[kk].x = Nm[arow * RR + kk * 4 + koff];
        aN[kk].y = Nm[arow * RR + kk * 4 + koff + 1];
    }
    float rs = res_scale[0];

    for (int ct = wave; ct < BD / 16; ct += 8) {
        int col = ct * 16 + lanelo;
        v8f acc = {};
#pragma unroll
        for (int kk = 0; kk < 8; ++kk) {
            v2f b; b.x = Uw[(kk * 4 + koff) * BD + col]; b.y = Uw[(kk * 4 + koff + 1) * BD + col];
            acc = __builtin_amdgcn_wmma_f32_16x16x4_f32(false, aM[kk], false, b, (short)0, acc, false, false);
        }
#pragma unroll
        for (int kk = 0; kk < 8; ++kk) {
            v2f b; b.x = Vw[(kk * 4 + koff) * BD + col]; b.y = Vw[(kk * 4 + koff + 1) * BD + col];
            acc = __builtin_amdgcn_wmma_f32_16x16x4_f32(false, aN[kk], false, b, (short)0, acc, false, false);
        }
        V8FU av; av.v = acc;
        float ls = lscale[col];
#pragma unroll
        for (int i = 0; i < 8; ++i) {
            long grow = row0 + i + hi * 8;
            X1[grow * BD + col] = X[grow * BD + col] + rs * g[grow] * av.f[i] * ls;
        }
    }
}

// ---------------------------------------------------------------- K4: fused FFN (bf16 WMMA + TDM stage)
#define HPAD 520   // padded ushort row stride (even -> dword-aligned pair loads)
__global__ __launch_bounds__(256) void k4_ffn(const float* __restrict__ X1,
                                              const unsigned short* __restrict__ W1b,
                                              const unsigned short* __restrict__ W2b,
                                              const float* __restrict__ b1,
                                              const float* __restrict__ b2,
                                              const float* __restrict__ fg,
                                              const float* __restrict__ fb,
                                              float* __restrict__ out) {
    __shared__ float xsh[32 * BD];              // staged X1 tile (f32, 64 KB) via TDM
    __shared__ unsigned short hsh[32 * HPAD];   // LN(X1) in bf16
    __shared__ unsigned short msh[32 * HPAD];   // gelu(h@W1^T+b1) in bf16
    int wave = threadIdx.x >> 5, lane = threadIdx.x & 31;
    long row0 = (long)blockIdx.x * 32;

    // warm the L2-resident bf16 weights (global_prefetch_b8)
    __builtin_prefetch(&W1b[(threadIdx.x * 1024) & (FF * BD - 1)], 0, 1);
    __builtin_prefetch(&W2b[(threadIdx.x * 1024) & (FF * BD - 1)], 0, 1);

    // ---- stage X1[row0:row0+32, :] -> LDS ----
#if __has_builtin(__builtin_amdgcn_tensor_load_to_lds) && __has_builtin(__builtin_amdgcn_s_wait_tensorcnt)
    if (threadIdx.x < 32) {   // wave 0 issues one 2-D TDM descriptor (EXEC-independent DMA)
        unsigned lds_off = (unsigned)(unsigned long long)(uintptr_t)xsh;
        unsigned long long ga = (unsigned long long)(uintptr_t)(X1 + row0 * BD);
        uint32x4 g0;
        g0[0] = 1u;                                   // count=1, is_restore=0, gather=0
        g0[1] = lds_off;                              // lds_addr (bytes)
        g0[2] = (unsigned)(ga & 0xFFFFFFFFu);         // global_addr[31:0]
        g0[3] = (unsigned)((ga >> 32) & 0x1FFFFFFu) | 0x80000000u;  // addr[56:32] | type=2
        int32x8 g1;
        g1[0] = 0x00020000;                 // data_size=2 (4 bytes); no multicast/pad/iterate
        g1[1] = (int)(BD << 16);            // tensor_dim0[15:0]=512 in bits 63:48
        g1[2] = (int)0x80000000u;           // tensor_dim0 hi=0 | tensor_dim1[15:0]=32768
        g1[3] = (int)(BD << 16);            // tensor_dim1 hi=0 | tile_dim0=512
        g1[4] = 32;                         // tile_dim1=32, tile_dim2=0
        g1[5] = BD;                         // tensor_dim0_stride[31:0]=512
        g1[6] = 0;                          // stride0 hi | dim1_stride lo
        g1[7] = 0;
        int32x4 z4 = {0, 0, 0, 0};
#if defined(__clang_major__) && (__clang_major__ >= 23)
        int32x8 z8 = {0, 0, 0, 0, 0, 0, 0, 0};
        __builtin_amdgcn_tensor_load_to_lds(g0, g1, z4, z4, z8, 0);
#else
        __builtin_amdgcn_tensor_load_to_lds(g0, g1, z4, z4, 0);
#endif
        __builtin_amdgcn_s_wait_tensorcnt(0);
    }
#else
    for (int idx = threadIdx.x; idx < 32 * BD; idx += 256)
        xsh[idx] = X1[row0 * BD + idx];
#endif
    __syncthreads();

    // phase 1: layernorm (from LDS stage) -> bf16 LDS (each wave handles 4 rows)
    for (int rr = 0; rr < 4; ++rr) {
        int r = wave * 4 + rr;
        float x[16]; float s = 0.f;
#pragma unroll
        for (int j = 0; j < 16; ++j) { x[j] = xsh[r * BD + lane + 32 * j]; s += x[j]; }
#pragma unroll
        for (int m = 16; m; m >>= 1) s += __shfl_xor(s, m, 32);
        float mean = s * (1.0f / BD);
        float v = 0.f;
#pragma unroll
        for (int j = 0; j < 16; ++j) { float t = x[j] - mean; v += t * t; }
#pragma unroll
        for (int m = 16; m; m >>= 1) v += __shfl_xor(v, m, 32);
        float rstd = rsqrtf(v * (1.0f / BD) + 1e-5f);
#pragma unroll
        for (int j = 0; j < 16; ++j) {
            int dd = lane + 32 * j;
            hsh[r * HPAD + dd] = f2bf((x[j] - mean) * rstd * fg[dd] + fb[dd]);
        }
    }
    __syncthreads();

    int hi = lane >> 4, lanelo = lane & 15;

    // phase 2: mid = gelu(h @ W1^T + b1)  -- 64 16x16 tiles over 8 waves
    for (int idx = wave; idx < 64; idx += 8) {
        int mt = idx >> 5, ct = idx & 31;
        int arow = mt * 16 + lanelo;
        int ncol = ct * 16 + lanelo;
        v8f acc = {};
        for (int kk = 0; kk < 16; ++kk) {
            V16U8 a, b;
#pragma unroll
            for (int j = 0; j < 8; ++j) {
                // ISA 16-bit A 16x32 layout: pairs at {0..7|16..23} (+8 for lanes 16-31)
                int kb = kk * 32 + ((j & 4) ? 16 : 0) + hi * 8 + 2 * (j & 3);
                a.u[j] = *(const unsigned int*)&hsh[arow * HPAD + kb];
                b.u[j] = *(const unsigned int*)&W1b[ncol * BD + kb];
            }
            acc = __builtin_amdgcn_wmma_f32_16x16x32_bf16(false, a.v, false, b.v, (short)0, acc, false, false);
        }
        V8FU av; av.v = acc;
#pragma unroll
        for (int i = 0; i < 8; ++i) {
            int m = mt * 16 + i + hi * 8;
            float v = av.f[i] + b1[ncol];
            msh[m * HPAD + ncol] = f2bf(0.5f * v * (1.0f + erff(v * 0.70710678118f)));
        }
    }
    __syncthreads();

    // phase 3: out = mid @ W2^T + b2 + X1 (residual from LDS stage)
    for (int idx = wave; idx < 64; idx += 8) {
        int mt = idx >> 5, ct = idx & 31;
        int arow = mt * 16 + lanelo;
        int ncol = ct * 16 + lanelo;
        v8f acc = {};
        for (int kk = 0; kk < 16; ++kk) {
            V16U8 a, b;
#pragma unroll
            for (int j = 0; j < 8; ++j) {
                int kb = kk * 32 + ((j & 4) ? 16 : 0) + hi * 8 + 2 * (j & 3);
                a.u[j] = *(const unsigned int*)&msh[arow * HPAD + kb];
                b.u[j] = *(const unsigned int*)&W2b[ncol * FF + kb];
            }
            acc = __builtin_amdgcn_wmma_f32_16x16x32_bf16(false, a.v, false, b.v, (short)0, acc, false, false);
        }
        V8FU av; av.v = acc;
#pragma unroll
        for (int i = 0; i < 8; ++i) {
            int m = mt * 16 + i + hi * 8;
            long grow = row0 + m;
            out[grow * BD + ncol] = av.f[i] + b2[ncol] + xsh[m * BD + ncol];
        }
    }
}

// ---------------------------------------------------------------- host launcher
extern "C" void kernel_launch(void* const* d_in, const int* in_sizes, int n_in,
                              void* d_out, int out_size, void* d_ws, size_t ws_size,
                              hipStream_t stream) {
    const float* X   = (const float*)d_in[0];
    const float* Uw  = (const float*)d_in[1];
    const float* Vw  = (const float*)d_in[2];
    const float* Aw  = (const float*)d_in[3];
    const float* Ab  = (const float*)d_in[4];
    const float* Bw  = (const float*)d_in[5];
    const float* Bb  = (const float*)d_in[6];
    const float* cv  = (const float*)d_in[7];
    const float* gw  = (const float*)d_in[8];
    const float* gb  = (const float*)d_in[9];
    const float* rsc = (const float*)d_in[10];
    const float* lsc = (const float*)d_in[11];
    const float* lng = (const float*)d_in[12];
    const float* lnb = (const float*)d_in[13];
    const float* fg  = (const float*)d_in[14];
    const float* fb  = (const float*)d_in[15];
    const float* W1  = (const float*)d_in[16];
    const float* b1  = (const float*)d_in[17];
    const float* W2  = (const float*)d_in[18];
    const float* b2  = (const float*)d_in[19];
    float* out = (float*)d_out;

    // workspace bump allocation (floats)
    float* ws = (float*)d_ws;
    const long NR = (long)BL * RR;           // 1,048,576
    float* P    = ws;
    float* S    = ws + 1 * NR;
    float* Q    = ws + 2 * NR;
    float* M    = ws + 3 * NR;
    float* Msuf = ws + 4 * NR;
    float* N    = ws + 5 * NR;
    float* cj   = ws + 6 * NR;               // BL*3
    float* g    = cj + (long)BL * 3;         // BL
    float* X1   = g + BL;                    // BL*BD
    unsigned short* W1b = (unsigned short*)(X1 + (long)BL * BD);
    unsigned short* W2b = W1b + (long)FF * BD;

    k0_cvt   <<<(FF * BD) / 256, 256, 0, stream>>>(W1, W2, W1b, W2b);
    k1_ln_proj<<<BL / 8, 256, 0, stream>>>(X, Uw, Vw, gw, lng, lnb, P, Q, cj);
    k2a_cumsum<<<BSZ, RR, 0, stream>>>(P, S);
    k2b_gmn  <<<BL / 256, 256, 0, stream>>>(S, Q, Aw, Ab, Bw, Bb, cv, M, N);
    k2c_sufsum<<<BSZ, RR, 0, stream>>>(M, Msuf);
    k2g_gate <<<BL / 256, 256, 0, stream>>>(cj, gb, g);
    k3_grad_x1<<<BL / 16, 256, 0, stream>>>(Msuf, N, Uw, Vw, X, g, rsc, lsc, X1);
    k4_ffn   <<<BL / 32, 256, 0, stream>>>(X1, W1b, W2b, b1, b2, fg, fb, out);
}
